// MBartMoEDecoderLayer_68839735820315
// MI455X (gfx1250) — matrix operand order repaired and checked by reading, not compiled, not run on hardware
//
#include <hip/hip_runtime.h>
#include <hip/hip_bf16.h>

// ---------------------------------------------------------------------------
// MBart MoE decoder layer for MI455X (gfx1250, wave32, WMMA).
// Shapes (fixed by the reference): B=4, S=256, SK=512, D=1024, NH=16, NKV=4,
// HD=64, DE=4096, NE=8. All GEMM dims are multiples of the 128x128x32 tile.
// ---------------------------------------------------------------------------

typedef __attribute__((ext_vector_type(16))) __bf16 bf16x16;
typedef __attribute__((ext_vector_type(8)))  __bf16 bf16x8;
typedef __attribute__((ext_vector_type(4)))  __bf16 bf16x4;
typedef __attribute__((ext_vector_type(8)))  float  f32x8;

// ---------------------------------------------------------------------------
// bf16 WMMA GEMM:  C = alpha*(A[M,K] @ W[K,N] + bias) [*gelu(gate)] [+residual]
// or, if accumulate: C += coef[b] * (...). coefB!=null enables per-batch
// early-exit (routed-out MoE experts skip entire tiles).
// Block = 256 threads = 8 waves; block tile 128x128; wave tile 32x64
// (2x4 WMMA 16x16x32 tiles). Global->reg->LDS software pipeline hides HBM
// latency behind the WMMA work of the previous K-step.
// ---------------------------------------------------------------------------
#define BM 128
#define BN 128
#define BK 32
#define LDS_STRIDE 40   // 32 + 8 halves padding (80B rows, conflict-free frags)

__global__ __launch_bounds__(256)
void gemm_wmma_bf16(const float* __restrict__ A, int lda,
                    const float* __restrict__ W,      // [K,N] row-major
                    const float* __restrict__ bias,   // [N] or null
                    const float* residual,            // [M,N] or null
                    const float* gate,                // [M,N] or null (may alias C)
                    const float* __restrict__ coefB,  // [num_batches] or null
                    int rowsPerBatch, float alpha, int accumulate,
                    float* C, int M, int N, int K)
{
    __shared__ __bf16 As[BM][LDS_STRIDE];
    __shared__ __bf16 Bt[BN][LDS_STRIDE];   // W tile stored transposed [n][k]

    const int tid    = threadIdx.x;
    const int lane   = tid & 31;
    const int wid    = tid >> 5;
    const int half   = lane >> 4;   // lane 0-15 vs 16-31 (ISA 16x16 layouts)
    const int r      = lane & 15;
    const int waveM  = wid & 3;     // 0..3, 32 rows each
    const int waveN  = wid >> 2;    // 0..1, 64 cols each
    const int blockM = blockIdx.y * BM;
    const int blockN = blockIdx.x * BN;

    float cf = 1.0f;
    if (coefB) {
        cf = coefB[blockM / rowsPerBatch];
        if (cf == 0.0f) return;     // expert not routed to this batch: skip
    }

    // staging coordinates (per thread, fixed across K loop)
    const int aRow = tid >> 3;            // +32 per chunk i
    const int aCol = (tid & 7) * 4;
    const int wRow = tid >> 5;            // +8 per chunk i
    const int wCol = (tid & 31) * 4;
    const float* aPtr = A + (size_t)(blockM + aRow) * lda + aCol;
    const float* wPtr = W + (size_t)wRow * N + blockN + wCol;

    float4 aReg[4], wReg[4];
    auto loadTile = [&](int kb) {
        #pragma unroll
        for (int i = 0; i < 4; ++i)
            aReg[i] = *(const float4*)(aPtr + (size_t)i * 32 * lda + kb);
        #pragma unroll
        for (int i = 0; i < 4; ++i)
            wReg[i] = *(const float4*)(wPtr + (size_t)(kb + i * 8) * N);
    };

    f32x8 acc[2][4] = {};
    loadTile(0);

    for (int kb = 0; kb < K; kb += BK) {
        // ---- publish current tile to LDS (native bf16 converts) ----
        #pragma unroll
        for (int i = 0; i < 4; ++i) {
            bf16x4 t;
            t[0] = (__bf16)aReg[i].x; t[1] = (__bf16)aReg[i].y;
            t[2] = (__bf16)aReg[i].z; t[3] = (__bf16)aReg[i].w;
            *(bf16x4*)&As[aRow + i * 32][aCol] = t;
        }
        #pragma unroll
        for (int i = 0; i < 4; ++i) {
            Bt[wCol + 0][wRow + i * 8] = (__bf16)wReg[i].x;
            Bt[wCol + 1][wRow + i * 8] = (__bf16)wReg[i].y;
            Bt[wCol + 2][wRow + i * 8] = (__bf16)wReg[i].z;
            Bt[wCol + 3][wRow + i * 8] = (__bf16)wReg[i].w;
        }
        __syncthreads();

        // ---- start next tile's global loads (overlap with WMMA below) ----
        if (kb + BK < K) loadTile(kb + BK);
        if (kb + 2 * BK < K) {      // warm L2 for the tile after next
            __builtin_prefetch(aPtr + (size_t)(kb + 2 * BK), 0, 1);
            __builtin_prefetch(wPtr + (size_t)(kb + 2 * BK) * N, 0, 1);
        }

        // ---- load fragments per ISA 16x16x32 bf16 VGPR layout ----
        bf16x16 afrag[2], bfrag[4];
        #pragma unroll
        for (int t = 0; t < 2; ++t) {
            int m = waveM * 32 + t * 16 + r;
            bf16x8 lo = *(const bf16x8*)&As[m][half * 8];        // K half*8..+7
            bf16x8 hi = *(const bf16x8*)&As[m][16 + half * 8];   // K 16+half*8..
            afrag[t] = __builtin_shufflevector(lo, hi,
                0,1,2,3,4,5,6,7,8,9,10,11,12,13,14,15);
        }
        #pragma unroll
        for (int t = 0; t < 4; ++t) {
            int n = waveN * 64 + t * 16 + r;
            bf16x8 lo = *(const bf16x8*)&Bt[n][half * 8];        // K half*8..+7
            bf16x8 hi = *(const bf16x8*)&Bt[n][16 + half * 8];
            bfrag[t] = __builtin_shufflevector(lo, hi,
                0,1,2,3,4,5,6,7,8,9,10,11,12,13,14,15);
        }
        #pragma unroll
        for (int ta = 0; ta < 2; ++ta)
            #pragma unroll
            for (int tb = 0; tb < 4; ++tb)
                acc[ta][tb] = __builtin_amdgcn_wmma_f32_16x16x32_bf16(
                    false, afrag[ta], false, bfrag[tb],
                    (short)0, acc[ta][tb], false, false);
        __syncthreads();
    }

    // ---- epilogue (C/D layout: VGPR v -> row v+8*half, lane r -> col) ----
    #pragma unroll
    for (int ta = 0; ta < 2; ++ta) {
        #pragma unroll
        for (int tb = 0; tb < 4; ++tb) {
            const int gn  = blockN + waveN * 64 + tb * 16 + r;
            const float bsv = bias ? bias[gn] : 0.0f;
            #pragma unroll
            for (int v = 0; v < 8; ++v) {
                const int gm = blockM + waveM * 32 + ta * 16 + half * 8 + v;
                const size_t idx = (size_t)gm * N + gn;
                float val = (acc[ta][tb][v] + bsv) * alpha;
                if (gate) {   // h = gelu(x@w1) * (x@w3)   (exact gelu)
                    float g = gate[idx];
                    val *= 0.5f * g * (1.0f + erff(g * 0.70710678118654752f));
                }
                if (residual) val += residual[idx];
                if (accumulate) C[idx] += cf * val;
                else            C[idx]  = val;
            }
        }
    }
}

// ---------------------------------------------------------------------------
// LayerNorm: one block per row of D=1024.
// ---------------------------------------------------------------------------
__global__ __launch_bounds__(256)
void ln_kernel(const float* __restrict__ x, const float* __restrict__ g,
               const float* __restrict__ b, float* __restrict__ y, int D)
{
    __shared__ float red0[256], red1[256];
    const int row = blockIdx.x;
    const float* xr = x + (size_t)row * D;
    float s1 = 0.f, s2 = 0.f;
    for (int j = threadIdx.x; j < D; j += 256) { float v = xr[j]; s1 += v; s2 += v * v; }
    red0[threadIdx.x] = s1; red1[threadIdx.x] = s2;
    __syncthreads();
    for (int off = 128; off > 0; off >>= 1) {
        if (threadIdx.x < off) {
            red0[threadIdx.x] += red0[threadIdx.x + off];
            red1[threadIdx.x] += red1[threadIdx.x + off];
        }
        __syncthreads();
    }
    const float mean = red0[0] / (float)D;
    const float var  = red1[0] / (float)D - mean * mean;
    const float rstd = rsqrtf(var + 1e-5f);
    float* yr = y + (size_t)row * D;
    for (int j = threadIdx.x; j < D; j += 256)
        yr[j] = (xr[j] - mean) * rstd * g[j] + b[j];
}

// ---------------------------------------------------------------------------
// GQA attention: one block per (batch, head). K/V for the head's kv-group are
// staged into (dynamic) LDS — 128KB self / 256KB cross, well under the 320KB
// WGP LDS. Each of 256 threads owns one query row; online softmax in regs.
// ---------------------------------------------------------------------------
__global__ __launch_bounds__(256)
void attention_kernel(const float* __restrict__ Q,   // [B,S,1024]
                      const float* __restrict__ K,   // [B,Skv,256]
                      const float* __restrict__ V,   // [B,Skv,256]
                      const float* __restrict__ mask,// [B,1,S,Skv]
                      float* __restrict__ O,         // [B,S,1024]
                      int S, int Skv)
{
    extern __shared__ float sm[];
    float* Ks = sm;
    float* Vs = sm + (size_t)Skv * 64;

    const int b   = blockIdx.x >> 4;   // NH = 16
    const int h   = blockIdx.x & 15;
    const int kvh = h >> 2;            // rep = NH/NKV = 4

    for (int idx = threadIdx.x; idx < Skv * 64; idx += 256) {
        const int kk = idx >> 6, dd = idx & 63;
        Ks[idx] = K[((size_t)b * Skv + kk) * 256 + kvh * 64 + dd];
        Vs[idx] = V[((size_t)b * Skv + kk) * 256 + kvh * 64 + dd];
    }
    __syncthreads();

    const int q = threadIdx.x;         // S == 256 == blockDim.x
    float qv[64];
    const float* qp = Q + ((size_t)b * S + q) * 1024 + h * 64;
    #pragma unroll
    for (int d = 0; d < 64; ++d) qv[d] = qp[d];

    float mx = -3.4e38f, l = 0.f;
    float o[64];
    #pragma unroll
    for (int d = 0; d < 64; ++d) o[d] = 0.f;

    const float* mrow = mask + ((size_t)b * S + q) * Skv;
    for (int k = 0; k < Skv; ++k) {
        const float* kr = Ks + (size_t)k * 64;
        float s = 0.f;
        #pragma unroll
        for (int d = 0; d < 64; ++d) s += qv[d] * kr[d];
        s += mrow[k];
        const float nm   = fmaxf(mx, s);
        const float corr = __expf(mx - nm);
        const float p    = __expf(s - nm);
        l = l * corr + p;
        const float* vr = Vs + (size_t)k * 64;
        #pragma unroll
        for (int d = 0; d < 64; ++d) o[d] = o[d] * corr + p * vr[d];
        mx = nm;
    }
    const float inv = 1.f / l;
    float* op = O + ((size_t)b * S + q) * 1024 + h * 64;
    #pragma unroll
    for (int d = 0; d < 64; ++d) op[d] = o[d] * inv;
}

// ---------------------------------------------------------------------------
// MoE routing coefficients: coef[e*B+b] = match(e,b) ? 1/#routed_codes : 0
// ---------------------------------------------------------------------------
__global__ void moe_coef_kernel(const int* __restrict__ langs, float* __restrict__ coef)
{
    if (threadIdx.x != 0 || blockIdx.x != 0) return;
    for (int b = 0; b < 4; ++b) {
        const int l0 = langs[b * 2], l1 = langs[b * 2 + 1];
        const int cnt = (l0 > 3) + (l1 > 3);
        const float rw = (cnt == 0) ? 1.0f : 1.0f / (float)cnt;
        for (int e = 0; e < 8; ++e) {
            const int code = 4 + e;
            coef[e * 4 + b] = ((l0 == code) || (l1 == code)) ? rw : 0.0f;
        }
    }
}

// ---------------------------------------------------------------------------
extern "C" void kernel_launch(void* const* d_in, const int* in_sizes, int n_in,
                              void* d_out, int out_size, void* d_ws, size_t ws_size,
                              hipStream_t stream)
{
    (void)in_sizes; (void)n_in; (void)out_size; (void)ws_size;

    const float* hidden = (const float*)d_in[0];
    const float* enc    = (const float*)d_in[1];
    const float* amask  = (const float*)d_in[2];
    const float* emask  = (const float*)d_in[3];
    const int*   langs  = (const int*)  d_in[4];
    const float* ln1_g  = (const float*)d_in[5];
    const float* ln1_b  = (const float*)d_in[6];
    const float* sa_q_w = (const float*)d_in[7];
    const float* sa_q_b = (const float*)d_in[8];
    const float* sa_k_w = (const float*)d_in[9];
    const float* sa_k_b = (const float*)d_in[10];
    const float* sa_v_w = (const float*)d_in[11];
    const float* sa_v_b = (const float*)d_in[12];
    const float* sa_o_w = (const float*)d_in[13];
    const float* sa_o_b = (const float*)d_in[14];
    const float* ln2_g  = (const float*)d_in[15];
    const float* ln2_b  = (const float*)d_in[16];
    const float* ca_q_w = (const float*)d_in[17];
    const float* ca_q_b = (const float*)d_in[18];
    const float* ca_k_w = (const float*)d_in[19];
    const float* ca_k_b = (const float*)d_in[20];
    const float* ca_v_w = (const float*)d_in[21];
    const float* ca_v_b = (const float*)d_in[22];
    const float* ca_o_w = (const float*)d_in[23];
    const float* ca_o_b = (const float*)d_in[24];
    const float* ln3_g  = (const float*)d_in[25];
    const float* ln3_b  = (const float*)d_in[26];
    const float* moe_w1 = (const float*)d_in[27];
    const float* moe_w2 = (const float*)d_in[28];
    const float* moe_w3 = (const float*)d_in[29];
    float* out = (float*)d_out;

    // workspace layout (floats): ~36 MB total
    float* ws = (float*)d_ws;
    size_t off = 0;
    float* xln  = ws + off; off += (size_t)1024 * 1024;
    float* qbuf = ws + off; off += (size_t)1024 * 1024;
    float* kbuf = ws + off; off += (size_t)2048 * 256;
    float* vbuf = ws + off; off += (size_t)2048 * 256;
    float* attn = ws + off; off += (size_t)1024 * 1024;
    float* h1   = ws + off; off += (size_t)1024 * 1024;
    float* a1   = ws + off; off += (size_t)1024 * 4096;
    float* coef = ws + off; off += 64;

    const dim3 blk(256);
    const float qscale = 0.125f;   // HD^-0.5

    auto gemm = [&](const float* A, const float* W, const float* bias,
                    const float* res, const float* gate, const float* cf,
                    float alpha, int accum, float* C, int M, int N, int K) {
        dim3 grid(N / BN, M / BM);
        gemm_wmma_bf16<<<grid, blk, 0, stream>>>(A, K, W, bias, res, gate, cf,
                                                 256, alpha, accum, C, M, N, K);
    };

    // ---- self attention (causal GQA, pre-LN) ----
    ln_kernel<<<1024, blk, 0, stream>>>(hidden, ln1_g, ln1_b, xln, 1024);
    gemm(xln, sa_q_w, sa_q_b, nullptr, nullptr, nullptr, qscale, 0, qbuf, 1024, 1024, 1024);
    gemm(xln, sa_k_w, sa_k_b, nullptr, nullptr, nullptr, 1.0f,   0, kbuf, 1024,  256, 1024);
    gemm(xln, sa_v_w, sa_v_b, nullptr, nullptr, nullptr, 1.0f,   0, vbuf, 1024,  256, 1024);
    attention_kernel<<<64, blk, 2 * 256 * 64 * sizeof(float), stream>>>(
        qbuf, kbuf, vbuf, amask, attn, 256, 256);
    gemm(attn, sa_o_w, sa_o_b, hidden, nullptr, nullptr, 1.0f, 0, h1, 1024, 1024, 1024);

    // ---- cross attention over encoder states ----
    ln_kernel<<<1024, blk, 0, stream>>>(h1, ln2_g, ln2_b, xln, 1024);
    gemm(xln, ca_q_w, ca_q_b, nullptr, nullptr, nullptr, qscale, 0, qbuf, 1024, 1024, 1024);
    gemm(enc, ca_k_w, ca_k_b, nullptr, nullptr, nullptr, 1.0f,   0, kbuf, 2048,  256, 1024);
    gemm(enc, ca_v_w, ca_v_b, nullptr, nullptr, nullptr, 1.0f,   0, vbuf, 2048,  256, 1024);
    attention_kernel<<<64, blk, 2 * 512 * 64 * sizeof(float), stream>>>(
        qbuf, kbuf, vbuf, emask, attn, 256, 512);
    // writes the residual base (r) straight into d_out
    gemm(attn, ca_o_w, ca_o_b, h1, nullptr, nullptr, 1.0f, 0, out, 1024, 1024, 1024);

    // ---- language-routed MoE FFN, accumulated into d_out ----
    ln_kernel<<<1024, blk, 0, stream>>>(out, ln3_g, ln3_b, xln, 1024);
    moe_coef_kernel<<<1, 32, 0, stream>>>(langs, coef);
    for (int e = 0; e < 8; ++e) {
        const float* w1 = moe_w1 + (size_t)e * 1024 * 4096;
        const float* w2 = moe_w2 + (size_t)e * 4096 * 1024;
        const float* w3 = moe_w3 + (size_t)e * 1024 * 4096;
        const float* cf = coef + e * 4;
        // a1 = x @ w1          (skipped entirely where coef==0)
        gemm(xln, w1, nullptr, nullptr, nullptr, cf, 1.0f, 0, a1, 1024, 4096, 1024);
        // a1 = gelu(a1) * (x @ w3)   (in-place gate)
        gemm(xln, w3, nullptr, nullptr, a1, cf, 1.0f, 0, a1, 1024, 4096, 1024);
        // out += coef[b] * (a1 @ w2)
        gemm(a1,  w2, nullptr, nullptr, nullptr, cf, 1.0f, 1, out, 1024, 1024, 4096);
    }
}
